// MAHA_42855183679950
// MI455X (gfx1250) — matrix operand order repaired and checked by reading, not compiled, not run on hardware
//
#include <hip/hip_runtime.h>

// ============================================================================
// MI455X (gfx1250, wave32) implementation of the Set-Transformer neural
// process reference. All dense linear layers use v_wmma_f32_16x16x32_f16
// (f32 accumulate) with an LDS-staged A tile; attention / layernorm /
// reductions are wave32 VALU with shuffle reductions.
// Workspace requirement ~192 MB (see layout below).
//
// Param leaves are assumed flattened in jax pytree order (dict keys sorted):
//   top: Cx, Cy, Tx, Ty, params
//   params: emb1, emb2, ln, mlp1, mlp2, r_ST, z_ST
//   lin{b,w}; ln{b,g}; mab{k,ln0,ln1,o,q,v}; isab{I,mab0,mab1}
//   st{S,isab1,isab2,out,pma,sab1,sab2}
// ============================================================================

typedef __attribute__((ext_vector_type(16))) _Float16 v16h;
typedef __attribute__((ext_vector_type(8)))  _Float16 h8;
typedef __attribute__((ext_vector_type(8)))  float    v8f;

#define BB    16
#define NC    4000
#define NT    2000
#define CNUM  10
#define DIN   512
#define DH    128
#define HEADS 4
#define NI    32
#define SEG   (NC / CNUM)       // 400
#define BNSEG (BB * CNUM)       // 160

#define ASTR  40                // LDS row stride in halfs (80B, bank-conflict free)

// ---------------------------------------------------------------------------
// GEMM: out[M,N] = act(A[M,K] (row stride lda) @ W[K,N] + bias[N]) (+ res)
// Block: 128 threads = 4 waves; block tile = 64 rows x 64 cols.
// Each wave: 4 M-tiles x 1 N-tile -> 4 WMMAs per K-step, B fragment reused 4x.
// A tile (64x32) cooperatively converted f32->f16 into LDS once per K-step,
// fragments pulled with ds_load_b128.
// Requires N % 64 == 0 (all N here are 128 or 256); any M (edges clamped).
// ---------------------------------------------------------------------------
__global__ void gemm_kernel(const float* __restrict__ A, int lda,
                            const float* __restrict__ W,
                            const float* __restrict__ bias,
                            const float* __restrict__ res,
                            int relu, float* __restrict__ out,
                            int M, int K, int N)
{
    __shared__ _Float16 sA[64 * ASTR];

    const int lane  = threadIdx.x & 31;
    const int wave  = threadIdx.x >> 5;
    const int hi    = lane >> 4;
    const int l16   = lane & 15;
    const int m0    = blockIdx.x * 64;
    const int ntile = blockIdx.y * 4 + wave;
    const int col   = ntile * 16 + l16;           // always < N (N % 64 == 0)

    v8f acc0 = {}, acc1 = {}, acc2 = {}, acc3 = {};

    for (int kk = 0; kk < K; kk += 32) {
        __syncthreads();                            // previous-iter reads done
        // ---- cooperative A-tile fill: 64 rows x 32 cols, f32 -> f16 -------
#pragma unroll
        for (int it = 0; it < 16; ++it) {
            int e   = threadIdx.x + it * 128;       // 0..2047
            int r   = e >> 5;
            int c   = e & 31;
            int ar  = m0 + r; if (ar >= M) ar = M - 1;
            sA[r * ASTR + c] = (_Float16)A[(size_t)ar * lda + kk + c];
        }
        __syncthreads();

        // ---- B fragment: lane holds col, halfs j -> K = kk + hi*16 + j ----
        v16h bf;
        const float* wp = W + (size_t)(kk + hi * 16) * N + col;
        __builtin_prefetch(wp + (size_t)32 * N, 0, 1);
#pragma unroll
        for (int j = 0; j < 16; ++j) bf[j] = (_Float16)wp[(size_t)j * N];

        // ---- A fragments from LDS + 4 WMMAs -------------------------------
        v16h a0, a1, a2, a3;
#pragma unroll
        for (int mt = 0; mt < 4; ++mt) {
            const _Float16* ab = &sA[(mt * 16 + l16) * ASTR + hi * 16];
            h8 lo = *(const h8*)ab;
            h8 hb = *(const h8*)(ab + 8);
            v16h a;
#pragma unroll
            for (int j = 0; j < 8; ++j) { a[j] = lo[j]; a[j + 8] = hb[j]; }
            if (mt == 0) a0 = a; else if (mt == 1) a1 = a;
            else if (mt == 2) a2 = a; else a3 = a;
        }
        acc0 = __builtin_amdgcn_wmma_f32_16x16x32_f16(false, a0, false, bf, (short)0, acc0, false, false);
        acc1 = __builtin_amdgcn_wmma_f32_16x16x32_f16(false, a1, false, bf, (short)0, acc1, false, false);
        acc2 = __builtin_amdgcn_wmma_f32_16x16x32_f16(false, a2, false, bf, (short)0, acc2, false, false);
        acc3 = __builtin_amdgcn_wmma_f32_16x16x32_f16(false, a3, false, bf, (short)0, acc3, false, false);
    }

    // ---- epilogue: D vgpr i, lane -> row = mtile + hi*8 + i, col ----------
    const float bcol = bias[col];
    const bool nactive = (ntile * 16) < N;
#pragma unroll
    for (int mt = 0; mt < 4; ++mt) {
        v8f acc = (mt == 0) ? acc0 : (mt == 1) ? acc1 : (mt == 2) ? acc2 : acc3;
        int rbase = m0 + mt * 16 + hi * 8;
        if (res) {
#pragma unroll
            for (int i = 0; i < 8; ++i) {
                int orow = rbase + i;
                if (nactive && orow < M) {
                    size_t idx = (size_t)orow * N + col;
                    float v = acc[i] + bcol;
                    if (relu) v = fmaxf(v, 0.f);
                    out[idx] = v + res[idx];
                }
            }
        } else {
#pragma unroll
            for (int i = 0; i < 8; ++i) {
                int orow = rbase + i;
                if (nactive && orow < M) {
                    size_t idx = (size_t)orow * N + col;
                    float v = acc[i] + bcol;
                    if (relu) v = fmaxf(v, 0.f);
                    out[idx] = v;
                }
            }
        }
    }
}

// M==1 fallback (PMA seed S @ Wq): plain VALU.
__global__ void linear1_kernel(const float* __restrict__ x,
                               const float* __restrict__ W,
                               const float* __restrict__ bias,
                               float* __restrict__ out, int K, int N)
{
    int j = blockIdx.x * blockDim.x + threadIdx.x;
    if (j >= N) return;
    float acc = bias[j];
    for (int k = 0; k < K; ++k) acc += x[k] * W[(size_t)k * N + j];
    out[j] = acc;
}

// ---------------------------------------------------------------------------
// Attention: o[b,row,h*32+d] = q + sum_m softmax(q.k/sqrt(128)) v
// One wave per (batch, head, query row); lane = head-dim element (dh = 32).
// Online (flash) softmax. qBatchStride==0 broadcasts q over the batch dim.
// ---------------------------------------------------------------------------
__global__ void attn_kernel(const float* __restrict__ q, int qBatchStride,
                            const float* __restrict__ k,
                            const float* __restrict__ v,
                            float* __restrict__ out,
                            int Bn, int n, int m, float scale)
{
    int wid  = blockIdx.x * (blockDim.x >> 5) + (threadIdx.x >> 5);
    int lane = threadIdx.x & 31;
    int total = Bn * HEADS * n;
    if (wid >= total) return;
    int row = wid % n;
    int t   = wid / n;
    int h   = t % HEADS;
    int b   = t / HEADS;

    float qd = q[(size_t)b * qBatchStride + (size_t)row * DH + h * 32 + lane];
    const float* kb = k + (size_t)b * m * DH + h * 32;
    const float* vb = v + (size_t)b * m * DH + h * 32;

    float runmax = -1e30f, runsum = 0.f, acc = 0.f;
    for (int j = 0; j < m; ++j) {
        float p = qd * kb[(size_t)j * DH + lane];
#pragma unroll
        for (int o = 16; o > 0; o >>= 1) p += __shfl_xor(p, o);
        float s  = p * scale;
        float nm = fmaxf(runmax, s);
        float cr = __expf(runmax - nm);
        float e  = __expf(s - nm);
        acc    = acc * cr + e * vb[(size_t)j * DH + lane];
        runsum = runsum * cr + e;
        runmax = nm;
    }
    out[(size_t)(b * n + row) * DH + h * 32 + lane] = qd + acc / runsum;
}

// LayerNorm over last dim (128). One wave per row, 4 elems/lane.
__global__ void ln_kernel(const float* __restrict__ in, float* __restrict__ out,
                          const float* __restrict__ g, const float* __restrict__ b,
                          int rows)
{
    int w = blockIdx.x * (blockDim.x >> 5) + (threadIdx.x >> 5);
    if (w >= rows) return;
    int lane = threadIdx.x & 31;
    const float* r = in + (size_t)w * DH;
    float v0 = r[lane], v1 = r[lane + 32], v2 = r[lane + 64], v3 = r[lane + 96];
    float s = v0 + v1 + v2 + v3;
#pragma unroll
    for (int o = 16; o > 0; o >>= 1) s += __shfl_xor(s, o);
    float mean = s * (1.f / DH);
    float d0 = v0 - mean, d1 = v1 - mean, d2 = v2 - mean, d3 = v3 - mean;
    float qs = d0 * d0 + d1 * d1 + d2 * d2 + d3 * d3;
#pragma unroll
    for (int o = 16; o > 0; o >>= 1) qs += __shfl_xor(qs, o);
    float inv = rsqrtf(qs * (1.f / DH) + 1e-5f);
    float* o = out + (size_t)w * DH;
    o[lane]      = d0 * inv * g[lane]      + b[lane];
    o[lane + 32] = d1 * inv * g[lane + 32] + b[lane + 32];
    o[lane + 64] = d2 * inv * g[lane + 64] + b[lane + 64];
    o[lane + 96] = d3 * inv * g[lane + 96] + b[lane + 96];
}

// argmax of one-hot rows (first-occurrence ties, matches jnp.argmax)
__global__ void argmax_kernel(const float* __restrict__ y, int* __restrict__ cls,
                              int rows, int c)
{
    int i = blockIdx.x * blockDim.x + threadIdx.x;
    if (i >= rows) return;
    const float* r = y + (size_t)i * c;
    int best = 0; float bv = r[0];
    for (int j = 1; j < c; ++j) if (r[j] > bv) { bv = r[j]; best = j; }
    cls[i] = best;
}

// Stable counting sort by class per batch (== stable argsort of labels).
__global__ void order_kernel(const int* __restrict__ cls, int* __restrict__ order,
                             int n, int c)
{
    if (threadIdx.x != 0) return;
    int b = blockIdx.x;
    int counts[16];
    for (int j = 0; j < c; ++j) counts[j] = 0;
    const int* cb = cls + (size_t)b * n;
    for (int i = 0; i < n; ++i) counts[cb[i]]++;
    int offs[16]; int run = 0;
    for (int j = 0; j < c; ++j) { offs[j] = run; run += counts[j]; }
    int* ob = order + (size_t)b * n;
    for (int i = 0; i < n; ++i) ob[offs[cb[i]]++] = i;
}

// Gather rows: dst[b,pos,:] = src[b,order[b,pos],:]
__global__ void gather_kernel(const float* __restrict__ src,
                              const int* __restrict__ order,
                              float* __restrict__ dst, int n)
{
    int r = blockIdx.x;                 // global sorted row (b*n+pos)
    int b = r / n;
    int srow = b * n + order[r];
    dst[(size_t)r * DH + threadIdx.x] = src[(size_t)srow * DH + threadIdx.x];
}

// Fused logits = t0 @ dec^T, log-softmax, NLL + accuracy accumulation.
__global__ void logits_nll_kernel(const float* __restrict__ t0,
                                  const float* __restrict__ dec,   // (B,C,128)
                                  const float* __restrict__ Ty,
                                  float* __restrict__ accum)
{
    __shared__ float sdec[CNUM * DH];
    __shared__ float red[256];
    int b = blockIdx.x;
    for (int i = threadIdx.x; i < CNUM * DH; i += blockDim.x)
        sdec[i] = dec[(size_t)b * CNUM * DH + i];
    __syncthreads();

    int t = blockIdx.y * blockDim.x + threadIdx.x;
    float nll = 0.f, corr = 0.f;
    if (t < NT) {
        const float* x = t0 + ((size_t)b * NT + t) * DH;
        float lg[CNUM];
#pragma unroll
        for (int c = 0; c < CNUM; ++c) lg[c] = 0.f;
        for (int k = 0; k < DH; ++k) {
            float xv = x[k];
#pragma unroll
            for (int c = 0; c < CNUM; ++c) lg[c] += xv * sdec[c * DH + k];
        }
        float mx = lg[0]; int am = 0;
#pragma unroll
        for (int c = 1; c < CNUM; ++c) if (lg[c] > mx) { mx = lg[c]; am = c; }
        float se = 0.f;
#pragma unroll
        for (int c = 0; c < CNUM; ++c) se += __expf(lg[c] - mx);
        float lse = mx + __logf(se);
        const float* ty = Ty + ((size_t)b * NT + t) * CNUM;
        int lab = 0; float bv = ty[0];
#pragma unroll
        for (int c = 1; c < CNUM; ++c) if (ty[c] > bv) { bv = ty[c]; lab = c; }
        nll  = lse - lg[lab];
        corr = (am == lab) ? 1.f : 0.f;
    }
    red[threadIdx.x] = nll; __syncthreads();
    for (int st = 128; st > 0; st >>= 1) {
        if (threadIdx.x < st) red[threadIdx.x] += red[threadIdx.x + st];
        __syncthreads();
    }
    if (threadIdx.x == 0) atomicAdd(&accum[0], red[0]);
    __syncthreads();
    red[threadIdx.x] = corr; __syncthreads();
    for (int st = 128; st > 0; st >>= 1) {
        if (threadIdx.x < st) red[threadIdx.x] += red[threadIdx.x + st];
        __syncthreads();
    }
    if (threadIdx.x == 0) atomicAdd(&accum[1], red[0]);
}

__global__ void zero2_kernel(float* a)
{
    if (threadIdx.x < 2) a[threadIdx.x] = 0.f;
}

__global__ void finalize_kernel(const float* __restrict__ accum, float* __restrict__ out)
{
    if (threadIdx.x == 0) {
        out[0] = accum[0] / (float)BB;                        // NLL: sum / B
        out[1] = accum[1] * 100.0f / ((float)NT * (float)BB); // accuracy %
    }
}

// ============================================================================
// Host-side orchestration
// ============================================================================
struct Lin { const float* b; const float* w; };
struct LNp { const float* b; const float* g; };
struct Mab { Lin k; LNp ln0, ln1; Lin o, q, v; };
struct Isab { const float* I; Mab mab0, mab1; };
struct ST { const float* S; Isab isab1, isab2; Lin out; Mab pma, sab1, sab2; };

static Lin takeLin(void* const* d, int& i) { Lin l; l.b = (const float*)d[i++]; l.w = (const float*)d[i++]; return l; }
static LNp takeLN (void* const* d, int& i) { LNp l; l.b = (const float*)d[i++]; l.g = (const float*)d[i++]; return l; }
static Mab takeMab(void* const* d, int& i) {
    Mab m; m.k = takeLin(d, i); m.ln0 = takeLN(d, i); m.ln1 = takeLN(d, i);
    m.o = takeLin(d, i); m.q = takeLin(d, i); m.v = takeLin(d, i); return m;
}
static Isab takeIsab(void* const* d, int& i) {
    Isab s; s.I = (const float*)d[i++]; s.mab0 = takeMab(d, i); s.mab1 = takeMab(d, i); return s;
}
static ST takeST(void* const* d, int& i) {
    ST s; s.S = (const float*)d[i++]; s.isab1 = takeIsab(d, i); s.isab2 = takeIsab(d, i);
    s.out = takeLin(d, i); s.pma = takeMab(d, i); s.sab1 = takeMab(d, i); s.sab2 = takeMab(d, i);
    return s;
}

static void gemm(hipStream_t s, const float* A, int lda, const Lin& L,
                 const float* res, int relu, float* out, int M, int K, int N)
{
    dim3 grid((M + 63) / 64, (N + 63) / 64);
    gemm_kernel<<<grid, 128, 0, s>>>(A, lda, L.w, L.b, res, relu, out, M, K, N);
}

static void lnorm(hipStream_t s, float* x, const LNp& p, int rows)
{
    ln_kernel<<<(rows + 7) / 8, 256, 0, s>>>(x, x, p.g, p.b, rows);
}

static const float kScale = 0.08838834764831845f; // 1/sqrt(128)

// Full MAB: out = ln1( o' + relu(o' Wo + bo) ), o' = ln0( q + attn(q,k,v) )
static void run_mab(hipStream_t s, const Mab& p,
                    const float* Qsrc, int Mq, int n, int qbs,
                    const float* Ksrc, int m, int Bn,
                    float* bq, float* bk, float* bv, float* bo, float* out)
{
    if (Mq == 1) linear1_kernel<<<1, 128, 0, s>>>(Qsrc, p.q.w, p.q.b, bq, DH, DH);
    else         gemm(s, Qsrc, DH, p.q, nullptr, 0, bq, Mq, DH, DH);
    gemm(s, Ksrc, DH, p.k, nullptr, 0, bk, Bn * m, DH, DH);
    gemm(s, Ksrc, DH, p.v, nullptr, 0, bv, Bn * m, DH, DH);
    int total = Bn * HEADS * n;
    attn_kernel<<<(total + 7) / 8, 256, 0, s>>>(bq, qbs, bk, bv, bo, Bn, n, m, kScale);
    lnorm(s, bo, p.ln0, Bn * n);
    gemm(s, bo, DH, p.o, bo, 1, out, Bn * n, DH, DH);
    lnorm(s, out, p.ln1, Bn * n);
}

// ISAB: H = mab0(I, X); out = mab1(X, H)
static void run_isab(hipStream_t s, const Isab& p, const float* X, int Bn, int n,
                     float* out, float* ga, float* gb,
                     float* sq, float* sk, float* so, float* sh)
{
    run_mab(s, p.mab0, p.I, NI, NI, 0, X, n, Bn, sq, ga, gb, so, sh);
    run_mab(s, p.mab1, X, Bn * n, n, n * DH, sh, NI, Bn, ga, sk, so, gb, out);
}

static void run_st(hipStream_t s, const ST& p, const float* X, int Bn, int n, int dout,
                   float* out, float* G0, float* G1, float* G2, float* G3,
                   float* sm0, float* sm1, float* sm2, float* sm3,
                   float* t0, float* t1, float* t2, float* t3, float* t4, float* t5,
                   float* sq)
{
    run_isab(s, p.isab1, X,  Bn, n, G0, G1, G2, sm0, sm1, sm2, sm3);
    run_isab(s, p.isab2, G0, Bn, n, G1, G2, G3, sm0, sm1, sm2, sm3);
    // PMA: Q = S (1x128 broadcast), K = h2 (G1)
    run_mab(s, p.pma, p.S, 1, 1, 0, G1, n, Bn, sq, G0, G2, t0, t1);
    // SAB1 / SAB2 on (Bn, 1, 128)
    run_mab(s, p.sab1, t1, Bn, 1, DH, t1, 1, Bn, t2, t3, t4, t5, t0);
    run_mab(s, p.sab2, t0, Bn, 1, DH, t0, 1, Bn, t2, t3, t4, t5, t1);
    gemm(s, t1, DH, p.out, nullptr, 0, out, Bn, DH, dout);
}

extern "C" void kernel_launch(void* const* d_in, const int* in_sizes, int n_in,
                              void* d_out, int out_size, void* d_ws, size_t ws_size,
                              hipStream_t stream)
{
    (void)in_sizes; (void)out_size; (void)ws_size;
    if (n_in < 190) return;

    const float* Cx = (const float*)d_in[0];
    const float* Cy = (const float*)d_in[1];
    const float* Tx = (const float*)d_in[2];
    const float* Ty = (const float*)d_in[3];

    int i = 4;
    Lin emb1 = takeLin(d_in, i);
    Lin emb2 = takeLin(d_in, i);
    LNp lnp  = takeLN (d_in, i);
    Lin mlp1 = takeLin(d_in, i);
    Lin mlp2 = takeLin(d_in, i);
    ST  rST  = takeST (d_in, i);
    ST  zST  = takeST (d_in, i);

    // ---------------- workspace layout (floats), ~192 MB total --------------
    const size_t BIG = (size_t)BNSEG * SEG * DH;   // 8,192,000
    size_t off = 0;
    float* base = (float*)d_ws;
    auto alloc = [&](size_t nf) { float* p = base + off; off += nf; return p; };
    float* t0buf = alloc((size_t)BB * NT * DH);    // 4,096,000 (persists)
    float* segs  = alloc(BIG);                     // persists for both STs
    float* G0 = alloc(BIG);
    float* G1 = alloc(BIG);
    float* G2 = alloc(BIG);
    float* G3 = alloc(BIG);
    float* sm0 = alloc((size_t)BNSEG * NI * DH);
    float* sm1 = alloc((size_t)BNSEG * NI * DH);
    float* sm2 = alloc((size_t)BNSEG * NI * DH);
    float* sm3 = alloc((size_t)BNSEG * NI * DH);
    float* tt0 = alloc(BNSEG * DH);
    float* tt1 = alloc(BNSEG * DH);
    float* tt2 = alloc(BNSEG * DH);
    float* tt3 = alloc(BNSEG * DH);
    float* tt4 = alloc(BNSEG * DH);
    float* tt5 = alloc(BNSEG * DH);
    float* rc     = alloc(BNSEG * DH);
    float* zparam = alloc(BNSEG * 2 * DH);
    float* sq     = alloc(DH);
    float* accum  = alloc(8);
    int* cls   = (int*)alloc(BB * NC);
    int* order = (int*)alloc(BB * NC);

    // ---------------- embedding: emb2(relu(emb1(x))) ------------------------
    gemm(stream, Cx, DIN, emb1, nullptr, 1, G2, BB * NC, DIN, DH);
    gemm(stream, G2, DH,  emb2, nullptr, 0, G3, BB * NC, DH,  DH);   // c0 -> G3
    gemm(stream, Tx, DIN, emb1, nullptr, 1, G2, BB * NT, DIN, DH);
    gemm(stream, G2, DH,  emb2, nullptr, 0, t0buf, BB * NT, DH, DH); // t0

    // ---------------- stable class sort + gather -> segs (160,400,128) ------
    argmax_kernel<<<(BB * NC + 255) / 256, 256, 0, stream>>>(Cy, cls, BB * NC, CNUM);
    order_kernel<<<BB, 32, 0, stream>>>(cls, order, NC, CNUM);
    gather_kernel<<<BB * NC, DH, 0, stream>>>(G3, order, segs, NC);

    // ---------------- two set transformers -----------------------------------
    run_st(stream, rST, segs, BNSEG, SEG, DH,     rc,
           G0, G1, G2, G3, sm0, sm1, sm2, sm3, tt0, tt1, tt2, tt3, tt4, tt5, sq);
    run_st(stream, zST, segs, BNSEG, SEG, 2 * DH, zparam,
           G0, G1, G2, G3, sm0, sm1, sm2, sm3, tt0, tt1, tt2, tt3, tt4, tt5, sq);

    // ---------------- decoder: dec = ln(r_c + mlp2(relu(mlp1(z_mean)))) -----
    gemm(stream, zparam, 2 * DH, mlp1, nullptr, 1, tt2, BNSEG, DH, DH); // z_mean slice via lda
    gemm(stream, tt2, DH, mlp2, rc, 0, tt3, BNSEG, DH, DH);
    lnorm(stream, tt3, lnp, BNSEG);                                     // dec (16,10,128)

    // ---------------- logits + log-softmax + NLL/accuracy -------------------
    zero2_kernel<<<1, 32, 0, stream>>>(accum);
    dim3 lg(BB, (NT + 255) / 256);
    logits_nll_kernel<<<lg, 256, 0, stream>>>(t0buf, tt3, Ty, accum);
    finalize_kernel<<<1, 32, 0, stream>>>(accum, (float*)d_out);
}
